// BlockCrossAttention_18476949307505
// MI455X (gfx1250) — matrix operand order repaired and checked by reading, not compiled, not run on hardware
//
#include <hip/hip_runtime.h>

typedef float  v8f  __attribute__((ext_vector_type(8)));
typedef __bf16 v16bf __attribute__((ext_vector_type(16)));
typedef unsigned short v8u  __attribute__((ext_vector_type(8)));
typedef unsigned short v16u __attribute__((ext_vector_type(16)));
typedef unsigned short ushort_t;

#define B_    2
#define LDEC  8192
#define LENC  4096
#define D_    1024
#define NB    512
#define KV_   4
#define HPG_  4
#define DH_   64
#define SCALE_ 0.125f   // 1/sqrt(64)

__device__ __forceinline__ unsigned short f2bf(float f) {
  unsigned u = __float_as_uint(f);
  unsigned r = u + 0x7fffu + ((u >> 16) & 1u);   // round to nearest even
  return (unsigned short)(r >> 16);
}

// A-side fragment: two adjacent 8-element chunks (k = 16*half + s) -> pack
__device__ __forceinline__ v16bf mk_frag(v8u lo, v8u hi) {
  v16u r;
#pragma unroll
  for (int i = 0; i < 8; ++i) { r[i] = lo[i]; r[8 + i] = hi[i]; }
  return __builtin_bit_cast(v16bf, r);
}

// B-side fragment: 16 contiguous bf16 (32 bytes, 32B-aligned at call sites)
__device__ __forceinline__ v16bf ld_frag(const ushort_t* p) {
  return __builtin_bit_cast(v16bf, *(const v16u*)p);
}

__device__ __forceinline__ v8f wmma_bf16(v16bf a, v16bf b, v8f c) {
  // D(16x16 f32) = A(16x32 bf16) * B(32x16 bf16) + C
  return __builtin_amdgcn_wmma_f32_16x16x32_bf16(false, a, false, b, (short)0, c,
                                                 false, false);
}

// -------- pooled[b,n,d] = mean of 16 decoder tokens, stored bf16 -----------
__global__ void pool_kernel(const float* __restrict__ x, ushort_t* __restrict__ pooled) {
  int idx = blockIdx.x * blockDim.x + threadIdx.x;   // B*NB*D = 1M
  int d  = idx & (D_ - 1);
  int bn = idx >> 10;
  int b  = bn >> 9;
  int n  = bn & (NB - 1);
  const float* xp = x + ((size_t)(b * LDEC + n * 16)) * D_ + d;
  float s = 0.f;
#pragma unroll
  for (int t = 0; t < 16; ++t) s += xp[(size_t)t * D_];
  pooled[idx] = f2bf(s * (1.0f / 16.0f));
}

// -------- elementwise f32 -> bf16 ------------------------------------------
__global__ void cvt_kernel(const float* __restrict__ src, ushort_t* __restrict__ dst, int n) {
  int idx = blockIdx.x * blockDim.x + threadIdx.x;
  if (idx < n) dst[idx] = f2bf(src[idx]);
}

// -------- weight convert + transpose: src[K,N] f32 -> dst[N,K] bf16 --------
__global__ void cvtT_kernel(const float* __restrict__ src, ushort_t* __restrict__ dst,
                            int K, int N) {
  int idx = blockIdx.x * blockDim.x + threadIdx.x;   // K*N
  int k = idx / N, n = idx - k * N;                  // coalesced read
  dst[(size_t)n * K + k] = f2bf(src[idx]);
}

// -------- V transpose (bf16): v[b,l,g,d] -> vt[b,g,d,l] --------------------
__global__ void vtrans_kernel(const ushort_t* __restrict__ vb, ushort_t* __restrict__ vt) {
  int idx = blockIdx.x * blockDim.x + threadIdx.x;   // B*KV*DH*LENC = 2M
  int l    = idx & (LENC - 1);
  int rest = idx >> 12;
  int d    = rest & (DH_ - 1);
  int gb   = rest >> 6;                              // gb = b*KV + g
  int g = gb & 3, b = gb >> 2;
  vt[idx] = vb[((size_t)(b * LENC + l) * KV_ + g) * DH_ + d];
}

// -------- bf16 WMMA GEMM: C[M,N] = A[M,K] * B[K,N] -------------------------
// A row-major bf16; BT = B^T row-major bf16 [N,K]. Wave owns a 16x64 strip.
template <bool OUT_BF16>
__global__ void gemm_bf16(const ushort_t* __restrict__ A, const ushort_t* __restrict__ BT,
                          void* __restrict__ Cv, int M, int N, int K) {
  int wv   = (blockIdx.x * blockDim.x + threadIdx.x) >> 5;
  int lane = threadIdx.x & 31;
  int half = lane >> 4;
  int l16  = lane & 15;
  int tilesN = N >> 6;
  int total  = (M >> 4) * tilesN;
  if (wv >= total) return;                 // wave-uniform: EXEC stays all-1
  int tm = wv / tilesN;
  int tn = wv - tm * tilesN;

  const ushort_t* arow  = A  + (size_t)(tm * 16 + l16) * K;             // A: M=lane&15
  const ushort_t* bbase = BT + (size_t)(tn * 64 + l16) * K + 16 * half; // B: N=lane&15
  v8f acc[4] = {};
  for (int k0 = 0; k0 < K; k0 += 32) {
    v16bf af = mk_frag(*(const v8u*)(arow + k0 + 16 * half),
                       *(const v8u*)(arow + k0 + 8 + 16 * half));
#pragma unroll
    for (int j = 0; j < 4; ++j) {
      acc[j] = wmma_bf16(af, ld_frag(bbase + (size_t)j * 16 * K + k0), acc[j]);
    }
  }
  // C/D layout: VGPR r -> row r (lanes 0-15) / r+8 (lanes 16-31), col lane&15
#pragma unroll
  for (int r = 0; r < 8; ++r) {
    size_t ro = (size_t)(tm * 16 + r + 8 * half) * N + tn * 64 + l16;
    if (OUT_BF16) {
      ushort_t* C = (ushort_t*)Cv;
      C[ro] = f2bf(acc[0][r]); C[ro + 16] = f2bf(acc[1][r]);
      C[ro + 32] = f2bf(acc[2][r]); C[ro + 48] = f2bf(acc[3][r]);
    } else {
      float* C = (float*)Cv;
      C[ro] = acc[0][r]; C[ro + 16] = acc[1][r];
      C[ro + 32] = acc[2][r]; C[ro + 48] = acc[3][r];
    }
  }
}

// -------- flash-style GQA cross attention (bf16 WMMA) ----------------------
// wave w -> (b, g, qh, 16-row n tile); 32 L-columns per chunk.
__global__ void __launch_bounds__(128)
attn_kernel(const ushort_t* __restrict__ q, const ushort_t* __restrict__ k,
            const ushort_t* __restrict__ vt, const int* __restrict__ amask,
            ushort_t* __restrict__ ctx) {
  __shared__ __align__(16) ushort_t plds[4 * 512];   // 16 rows x 32 cols per wave
  int wid  = threadIdx.x >> 5;
  int lane = threadIdx.x & 31;
  int half = lane >> 4;
  int l16  = lane & 15;
  int w  = blockIdx.x * 4 + wid;   // exactly 1024 waves
  int nt = w & 31;
  int qh = (w >> 5) & 3;
  int g  = (w >> 7) & 3;
  int b  = w >> 9;
  ushort_t* pw = plds + wid * 512;

  // Q fragments: A-layout, K-dim = d (two K=32 steps)
  int nrow = nt * 16 + l16;
  const ushort_t* qrow = q + (size_t)(((b * NB + nrow) * KV_ + g) * HPG_ + qh) * DH_;
  v16bf qa[2];
#pragma unroll
  for (int s = 0; s < 2; ++s) {
    int k0 = 32 * s;
    qa[s] = mk_frag(*(const v8u*)(qrow + k0 + 16 * half),
                    *(const v8u*)(qrow + k0 + 8 + 16 * half));
  }

  const ushort_t* kg  = k  + (size_t)b * LENC * (KV_ * DH_) + g * DH_;
  const ushort_t* vtg = vt + (size_t)(b * KV_ + g) * DH_ * LENC;
  const int* mk = amask + b * LENC;

  v8f acc[4] = {};
  float m[8], ssum[8];
#pragma unroll
  for (int r = 0; r < 8; ++r) { m[r] = -1e30f; ssum[r] = 0.f; }

  for (int lc = 0; lc < LENC / 32; ++lc) {
    int l0 = lc * 32;
    // ---- S tiles: Sa = cols l0..+15, Sb = cols l0+16..+31 ----
    v8f Sa = {}, Sb = {};
#pragma unroll
    for (int s = 0; s < 2; ++s) {
      int d0 = 32 * s;
      const ushort_t* kra = kg + (size_t)(l0 + l16) * (KV_ * DH_) + d0 + 16 * half;
      Sa = wmma_bf16(qa[s], ld_frag(kra), Sa);
      Sb = wmma_bf16(qa[s], ld_frag(kra + (size_t)16 * (KV_ * DH_)), Sb);
    }
    int mvA = mk[l0 + l16];
    int mvB = mk[l0 + 16 + l16];
#pragma unroll
    for (int i = 0; i < 8; ++i) {
      Sa[i] = (mvA == 0) ? -1e9f : Sa[i] * SCALE_;
      Sb[i] = (mvB == 0) ? -1e9f : Sb[i] * SCALE_;
    }

    // ---- online softmax over 32 columns; rows r / r+8 per half ----
    float plA[8], plB[8];
#pragma unroll
    for (int r = 0; r < 8; ++r) {
      float sa = Sa[r], sb = Sb[r];
      float rmax = fmaxf(sa, sb);
      rmax = fmaxf(rmax, __shfl_xor(rmax, 1, 16));
      rmax = fmaxf(rmax, __shfl_xor(rmax, 2, 16));
      rmax = fmaxf(rmax, __shfl_xor(rmax, 4, 16));
      rmax = fmaxf(rmax, __shfl_xor(rmax, 8, 16));
      float mnew = fmaxf(m[r], rmax);
      float corr = __expf(m[r] - mnew);
      float pa = __expf(sa - mnew);
      float pb = __expf(sb - mnew);
      float rsum = pa + pb;
      rsum += __shfl_xor(rsum, 1, 16);
      rsum += __shfl_xor(rsum, 2, 16);
      rsum += __shfl_xor(rsum, 4, 16);
      rsum += __shfl_xor(rsum, 8, 16);
      ssum[r] = ssum[r] * corr + rsum;
      m[r] = mnew;
      acc[0][r] *= corr; acc[1][r] *= corr; acc[2][r] *= corr; acc[3][r] *= corr;
      plA[r] = pa; plB[r] = pb;
    }

    // ---- stage P (bf16) C-layout -> row-major 16x32 in LDS ----
#pragma unroll
    for (int r = 0; r < 8; ++r) {
      int row = r + 8 * half;
      pw[row * 32 + l16]      = f2bf(plA[r]);
      pw[row * 32 + 16 + l16] = f2bf(plB[r]);
    }
    __syncthreads();

    // ---- O += P @ V : A = P (K=32 over l), B = V^T (l-contiguous) ----
    {
      const ushort_t* pr = pw + l16 * 32;
      v16bf pfrag = mk_frag(*(const v8u*)(pr + 16 * half),
                            *(const v8u*)(pr + 8 + 16 * half));
#pragma unroll
      for (int j = 0; j < 4; ++j) {
        const ushort_t* vr = vtg + (size_t)(j * 16 + l16) * LENC + l0 + 16 * half;
        acc[j] = wmma_bf16(pfrag, ld_frag(vr), acc[j]);
      }
    }
    __syncthreads();
  }

  // ---- normalize, store ctx bf16 [b, n, g, qh, d] ----
  const size_t obase = (size_t)(((b * NB + nt * 16) * KV_ + g) * HPG_ + qh) * DH_;
#pragma unroll
  for (int r = 0; r < 8; ++r) {
    int n = r + 8 * half;
    float inv = 1.0f / ssum[r];
    ushort_t* op = ctx + obase + (size_t)n * (KV_ * HPG_ * DH_) + l16;
    op[0]  = f2bf(acc[0][r] * inv);
    op[16] = f2bf(acc[1][r] * inv);
    op[32] = f2bf(acc[2][r] * inv);
    op[48] = f2bf(acc[3][r] * inv);
  }
}

// -------- broadcast block outputs back to token level ----------------------
__global__ void bcast_kernel(const float* __restrict__ outb, float* __restrict__ out) {
  size_t idx = (size_t)blockIdx.x * blockDim.x + threadIdx.x;  // B*LDEC*D
  int d = (int)(idx & (D_ - 1));
  size_t bt = idx >> 10;
  int t = (int)(bt & (LDEC - 1));
  int b = (int)(bt >> 13);
  out[idx] = outb[((size_t)(b * NB + (t >> 4)) << 10) + d];
}

extern "C" void kernel_launch(void* const* d_in, const int* in_sizes, int n_in,
                              void* d_out, int out_size, void* d_ws, size_t ws_size,
                              hipStream_t stream) {
  (void)in_sizes; (void)n_in; (void)out_size; (void)ws_size;
  const float* x   = (const float*)d_in[0];
  const float* enc = (const float*)d_in[1];
  const int*   msk = (const int*)d_in[2];
  const float* Wq  = (const float*)d_in[3];
  const float* Wk  = (const float*)d_in[4];
  const float* Wv  = (const float*)d_in[5];
  const float* Wo  = (const float*)d_in[6];
  float* out = (float*)d_out;

  // workspace layout: f32 outb first (alignment), then bf16 buffers (~44 MB)
  float* outb = (float*)d_ws;                               // 1M f32
  ushort_t* pooledb = (ushort_t*)(outb + 1048576);          // 1M
  ushort_t* encb = pooledb + 1048576;                       // 8M (B*LENC*D)
  ushort_t* qbb  = encb + (size_t)B_ * LENC * D_;           // 1M
  ushort_t* kbb  = qbb  + 1048576;                          // 2M
  ushort_t* vbb  = kbb  + 2097152;                          // 2M
  ushort_t* vtb  = vbb  + 2097152;                          // 2M
  ushort_t* ctxb = vtb  + 2097152;                          // 1M
  ushort_t* WqT  = ctxb + 1048576;                          // 1M
  ushort_t* WkT  = WqT  + 1048576;                          // 256K
  ushort_t* WvT  = WkT  + 262144;                           // 256K
  ushort_t* WoT  = WvT  + 262144;                           // 1M

  // prep: pooling + bf16 conversions (weights transposed for B-fragment loads)
  pool_kernel<<<(B_ * NB * D_) / 256, 256, 0, stream>>>(x, pooledb);
  cvt_kernel<<<(B_ * LENC * D_) / 256, 256, 0, stream>>>(enc, encb, B_ * LENC * D_);
  cvtT_kernel<<<(D_ * D_) / 256, 256, 0, stream>>>(Wq, WqT, D_, D_);
  cvtT_kernel<<<(D_ * KV_ * DH_) / 256, 256, 0, stream>>>(Wk, WkT, D_, KV_ * DH_);
  cvtT_kernel<<<(D_ * KV_ * DH_) / 256, 256, 0, stream>>>(Wv, WvT, D_, KV_ * DH_);
  cvtT_kernel<<<(D_ * D_) / 256, 256, 0, stream>>>(Wo, WoT, D_, D_);

  // projections (bf16 WMMA, f32 accumulate)
  gemm_bf16<true><<<256, 128, 0, stream>>>(pooledb, WqT, qbb, B_ * NB, D_, D_);
  gemm_bf16<true><<<512, 128, 0, stream>>>(encb, WkT, kbb, B_ * LENC, KV_ * DH_, D_);
  gemm_bf16<true><<<512, 128, 0, stream>>>(encb, WvT, vbb, B_ * LENC, KV_ * DH_, D_);
  vtrans_kernel<<<(B_ * KV_ * DH_ * LENC) / 256, 256, 0, stream>>>(vbb, vtb);

  // attention (1024 waves exactly)
  attn_kernel<<<256, 128, 0, stream>>>(qbb, kbb, vtb, msk, ctxb);

  // output projection (f32 out) + token broadcast
  gemm_bf16<false><<<256, 128, 0, stream>>>(ctxb, WoT, outb, B_ * NB, D_, D_);
  bcast_kernel<<<(int)(((size_t)B_ * LDEC * D_) / 256), 256, 0, stream>>>(outb, out);
}
